// TAEncoder_23295902613733
// MI455X (gfx1250) — matrix operand order repaired and checked by reading, not compiled, not run on hardware
//
#include <hip/hip_runtime.h>
#include <hip/hip_bf16.h>
#include <stdint.h>

// ---------------- problem constants (match reference) ----------------
#define N_NODES 50000
#define N_PAD   50016          // N rounded up to multiple of 32 (WMMA K-step)
#define E_EDGES 300000
#define F_IN    128
#define HID     256
#define G_GR    1000
#define NUM_GC  5
#define ALPHA_F 0.5f
#define EPS_LN  1e-5f
#define EPS_BN  1e-5f

typedef __attribute__((ext_vector_type(16))) __bf16 v16bf;
typedef __attribute__((ext_vector_type(8)))  float  v8f;
typedef unsigned short u16;
typedef unsigned int   u32;

union FragU { v16bf v; uint4 q[2]; };

// f32 -> bf16 round-to-nearest-even
__device__ __forceinline__ u16 f2bf(float f) {
    u32 x = __builtin_bit_cast(u32, f);
    u32 r = x + 0x7fffu + ((x >> 16) & 1u);
    return (u16)(r >> 16);
}

__device__ __forceinline__ float wave_sum(float v) {
#pragma unroll
    for (int o = 16; o > 0; o >>= 1) v += __shfl_xor(v, o, 32);
    return v;
}

// ---- WMMA fragment loaders (ISA 7.12.2, 16-bit 16x32 A / 32x16 B) ----
// A[M,K] row-major bf16. Lane L(0..15): row m0+L, K = {8h..8h+7, 16+8h..}
__device__ __forceinline__ v16bf load_frag_A(const u16* __restrict__ A, int lda,
                                             int m0, int k0) {
    int lane = threadIdx.x & 31;
    int r = lane & 15, half = lane >> 4;
    const u16* p = A + (size_t)(m0 + r) * lda + k0;
    FragU f;
    f.q[0] = *(const uint4*)(p + 8 * half);
    f.q[1] = *(const uint4*)(p + 16 + 8 * half);
    return f.v;
}
// B fragment from a [64 x K] row-major bf16 panel (B transposed layout).
// Lane: col (lane&15) of the 16-wide subtile, K = {8h..8h+7, 16+8h..}.
__device__ __forceinline__ v16bf load_frag_B(const u16* Bt, int ldb,
                                             int n0, int k0) {
    int lane = threadIdx.x & 31;
    int nn = lane & 15, half = lane >> 4;
    const u16* p = Bt + (size_t)(n0 + nn) * ldb + k0;
    FragU f;
    f.q[0] = *(const uint4*)(p + 8 * half);
    f.q[1] = *(const uint4*)(p + 16 + 8 * half);
    return f.v;
}

// =====================================================================
//  Generic bf16 WMMA GEMM: C[M,Ncol] = A[M,K] * Bt^T (+bias)
//  8 waves / block, wave = 16x64 output strip. M % 16 == 0, K <= 256.
//  B panel (64 cols x K) staged once per block in LDS (<= 32 KB);
//  A fragments software-pipelined one k-step ahead.
// =====================================================================
__global__ __launch_bounds__(256) void wmma_gemm(const u16* __restrict__ A,
                                                 const u16* __restrict__ Bt,
                                                 const float* __restrict__ bias,
                                                 float* __restrict__ C,
                                                 int M, int K, int Ncol) {
    __shared__ u16 sB[64 * 256];
    int n0 = blockIdx.x * 64;
    {   // cooperative copy: panel rows n0..n0+63 are contiguous (64*K elems)
        const uint4* src = (const uint4*)(Bt + (size_t)n0 * K);
        uint4* dst = (uint4*)sB;
        int chunks = (64 * K) >> 3;              // 8 bf16 per 16B chunk
        for (int i = threadIdx.x; i < chunks; i += 256) dst[i] = src[i];
    }
    __syncthreads();

    int wave = threadIdx.x >> 5;
    int m0 = (blockIdx.y * 8 + wave) * 16;
    if (m0 >= M) return;                         // wave-uniform tail guard

    v8f acc[4] = {};
    v16bf a = load_frag_A(A, K, m0, 0);          // pipelined A stream
    for (int k0 = 0; k0 < K; k0 += 32) {
        if (k0 + 128 < K)
            __builtin_prefetch(A + (size_t)m0 * K + k0 + 128, 0, 1);
        v16bf an = a;
        if (k0 + 32 < K) an = load_frag_A(A, K, m0, k0 + 32);
        v16bf b0 = load_frag_B(sB, K, 0,  k0);
        v16bf b1 = load_frag_B(sB, K, 16, k0);
        v16bf b2 = load_frag_B(sB, K, 32, k0);
        v16bf b3 = load_frag_B(sB, K, 48, k0);
        acc[0] = __builtin_amdgcn_wmma_f32_16x16x32_bf16(false, a, false, b0, (short)0, acc[0], false, false);
        acc[1] = __builtin_amdgcn_wmma_f32_16x16x32_bf16(false, a, false, b1, (short)0, acc[1], false, false);
        acc[2] = __builtin_amdgcn_wmma_f32_16x16x32_bf16(false, a, false, b2, (short)0, acc[2], false, false);
        acc[3] = __builtin_amdgcn_wmma_f32_16x16x32_bf16(false, a, false, b3, (short)0, acc[3], false, false);
        a = an;
    }
    int lane = threadIdx.x & 31, nn = lane & 15, half = lane >> 4;
#pragma unroll
    for (int j = 0; j < 4; ++j) {
        int col = n0 + 16 * j + nn;
        float bv = bias ? bias[col] : 0.f;
#pragma unroll
        for (int r = 0; r < 8; ++r)
            C[(size_t)(m0 + r + 8 * half) * Ncol + col] = acc[j][r] + bv;
    }
}

// =====================================================================
//  kvs = k^T v  : C[256,256], K = N_PAD (split-K across 8 waves + LDS red.)
//  kT,vT are [256, N_PAD] row-major bf16 (pre-transposed, zero-padded).
// =====================================================================
__global__ __launch_bounds__(256) void kvs_gemm(const u16* __restrict__ kT,
                                                const u16* __restrict__ vT,
                                                float* __restrict__ kvs,
                                                int npad) {
    int wave = threadIdx.x >> 5;
    int m0 = blockIdx.x * 16;
    int n0 = blockIdx.y * 64;

    v8f acc[4] = {};
    for (int k0 = wave * 32; k0 < npad; k0 += 32 * 8) {
        v16bf a = load_frag_A(kT, npad, m0, k0);
#pragma unroll
        for (int j = 0; j < 4; ++j) {
            v16bf b = load_frag_B(vT, npad, n0 + 16 * j, k0);
            acc[j] = __builtin_amdgcn_wmma_f32_16x16x32_bf16(false, a, false, b, (short)0, acc[j], false, false);
        }
    }
    __shared__ float red[8 * 1024];            // 8 waves x (16x64) = 32 KB
    int lane = threadIdx.x & 31, nn = lane & 15, half = lane >> 4;
#pragma unroll
    for (int j = 0; j < 4; ++j)
#pragma unroll
        for (int r = 0; r < 8; ++r)
            red[wave * 1024 + (r + 8 * half) * 64 + 16 * j + nn] = acc[j][r];
    __syncthreads();
    for (int idx = threadIdx.x; idx < 1024; idx += 256) {
        float s = 0.f;
#pragma unroll
        for (int w = 0; w < 8; ++w) s += red[w * 1024 + idx];
        int mm = idx >> 6, dd = idx & 63;
        kvs[(size_t)(m0 + mm) * HID + n0 + dd] = s;
    }
}

// ---------------------- prep / pointwise kernels ----------------------
__global__ void transpose_bf16(const float* __restrict__ Wsrc, u16* __restrict__ Wt,
                               int K, int Ncol) {
    int idx = blockIdx.x * blockDim.x + threadIdx.x;
    if (idx >= K * Ncol) return;
    int o = idx % Ncol, k = idx / Ncol;
    Wt[(size_t)o * K + k] = f2bf(Wsrc[(size_t)k * Ncol + o]);
}

__global__ void convert_bf16(const float* __restrict__ X, u16* __restrict__ Y, int n) {
    int i = blockIdx.x * blockDim.x + threadIdx.x;
    if (i < n) Y[i] = f2bf(X[i]);
}

__global__ void pack_bias3(const float* bq, const float* bk, const float* bv,
                           float* __restrict__ out) {
    int i = threadIdx.x + blockIdx.x * blockDim.x;
    if (i >= 768) return;
    out[i] = (i < 256) ? bq[i] : (i < 512) ? bk[i - 256] : bv[i - 512];
}

__global__ void zero_f32(float* p, int n) {
    int i = blockIdx.x * blockDim.x + threadIdx.x;
    if (i < n) p[i] = 0.f;
}

__global__ void init_deg(float* deg, int n) {
    int i = blockIdx.x * blockDim.x + threadIdx.x;
    if (i < n) deg[i] = 1.0f;                 // self loop
}

__global__ void edge_deg(const int* __restrict__ dst, float* __restrict__ deg, int e) {
    int i = blockIdx.x * blockDim.x + threadIdx.x;
    if (i < e) atomicAdd(&deg[dst[i]], 1.0f);
}

// LayerNorm(256) + ReLU, wave per row; writes f32 and bf16 copies
__global__ __launch_bounds__(256) void ln_relu_rows(const float* __restrict__ X,
                                                    const float* __restrict__ g,
                                                    const float* __restrict__ b,
                                                    float* __restrict__ Y,
                                                    u16* __restrict__ Yb, int nrows) {
    int row = blockIdx.x * 8 + (threadIdx.x >> 5);
    if (row >= nrows) return;
    int lane = threadIdx.x & 31;
    const float* xr = X + (size_t)row * HID;
    float v[8], s = 0.f;
#pragma unroll
    for (int i = 0; i < 8; ++i) { v[i] = xr[lane + 32 * i]; s += v[i]; }
    float mean = wave_sum(s) * (1.f / HID);
    float sq = 0.f;
#pragma unroll
    for (int i = 0; i < 8; ++i) { float d = v[i] - mean; sq += d * d; }
    float inv = rsqrtf(wave_sum(sq) * (1.f / HID) + EPS_LN);
#pragma unroll
    for (int i = 0; i < 8; ++i) {
        int c = lane + 32 * i;
        float y = fmaxf((v[i] - mean) * inv * g[c] + b[c], 0.f);
        Y[(size_t)row * HID + c]  = y;
        Yb[(size_t)row * HID + c] = f2bf(y);
    }
}

// QKV post: qb bf16, kT/vT transposed bf16 (+zero pad), ks_sum, ||q||^2, ||k||^2
__global__ __launch_bounds__(256) void qkv_post(const float* __restrict__ QKV,
                                                u16* __restrict__ qb,
                                                u16* __restrict__ kT,
                                                u16* __restrict__ vT,
                                                float* __restrict__ small,
                                                int n_nodes, int npad) {
    int n = blockIdx.x, c = threadIdx.x;
    if (n >= n_nodes) {                       // zero-pad tail columns for split-K
        kT[(size_t)c * npad + n] = 0;
        vT[(size_t)c * npad + n] = 0;
        return;
    }
    float q = QKV[(size_t)n * 768 + c];
    float k = QKV[(size_t)n * 768 + 256 + c];
    float v = QKV[(size_t)n * 768 + 512 + c];
    qb[(size_t)n * HID + c]  = f2bf(q);
    kT[(size_t)c * npad + n] = f2bf(k);
    vT[(size_t)c * npad + n] = f2bf(v);
    atomicAdd(&small[c], k);                  // ks_sum (raw)

    __shared__ float sq[8], sk[8];
    int lane = threadIdx.x & 31, w = threadIdx.x >> 5;
    float a = wave_sum(q * q), b2 = wave_sum(k * k);
    if (lane == 0) { sq[w] = a; sk[w] = b2; }
    __syncthreads();
    if (threadIdx.x == 0) {
        float A = 0.f, B = 0.f;
#pragma unroll
        for (int i = 0; i < 8; ++i) { A += sq[i]; B += sk[i]; }
        atomicAdd(&small[512], A);
        atomicAdd(&small[513], B);
    }
}

// Fold scale 1/(|q||k|) into kvs (transposed bf16 for GEMM B) and ks_sum
__global__ void kvs_prep(const float* __restrict__ kvs, float* __restrict__ small,
                         u16* __restrict__ kvsT) {
    int m = blockIdx.x, d = threadIdx.x;
    float s = rsqrtf(small[512]) * rsqrtf(small[513]);
    kvsT[(size_t)d * HID + m] = f2bf(kvs[(size_t)m * HID + d] * s);
    if (d == 0) small[256 + m] = small[m] * s;   // scaled ks_sum
}

// attn epilogue: norm, +n*v, residual, LN1, ReLU, row L2 -> x_trans
__global__ __launch_bounds__(256) void attn_epilogue(const float* __restrict__ T,
                                                     const float* __restrict__ QKV,
                                                     const float* __restrict__ H,
                                                     const float* __restrict__ small,
                                                     const float* __restrict__ g1,
                                                     const float* __restrict__ b1,
                                                     float* __restrict__ xt,
                                                     int nrows) {
    int row = blockIdx.x * 8 + (threadIdx.x >> 5);
    if (row >= nrows) return;
    int lane = threadIdx.x & 31;
    const float* ksum = small + 256;
    float q[8], vv[8], t[8], h[8], dot = 0.f;
#pragma unroll
    for (int i = 0; i < 8; ++i) {
        int c = lane + 32 * i;
        q[i]  = QKV[(size_t)row * 768 + c];
        vv[i] = QKV[(size_t)row * 768 + 512 + c];
        t[i]  = T[(size_t)row * HID + c];
        h[i]  = H[(size_t)row * HID + c];
        dot  += q[i] * ksum[c];
    }
    float norm = wave_sum(dot) + (float)N_NODES;
    float h1[8], s = 0.f;
#pragma unroll
    for (int i = 0; i < 8; ++i) {
        float attn = (t[i] + (float)N_NODES * vv[i]) / norm;
        h1[i] = ALPHA_F * attn + (1.f - ALPHA_F) * h[i];
        s += h1[i];
    }
    float mean = wave_sum(s) * (1.f / HID);
    float sq = 0.f;
#pragma unroll
    for (int i = 0; i < 8; ++i) { float d = h1[i] - mean; sq += d * d; }
    float inv = rsqrtf(wave_sum(sq) * (1.f / HID) + EPS_LN);
    float y[8], ss = 0.f;
#pragma unroll
    for (int i = 0; i < 8; ++i) {
        int c = lane + 32 * i;
        y[i] = fmaxf((h1[i] - mean) * inv * g1[c] + b1[c], 0.f);
        ss += y[i] * y[i];
    }
    float nrm = fmaxf(sqrtf(wave_sum(ss)), 1e-12f);
#pragma unroll
    for (int i = 0; i < 8; ++i)
        xt[(size_t)row * HID + lane + 32 * i] = y[i] / nrm;
}

// GCN: self-loop contribution + bias init
__global__ void gcn_self_init(const float* __restrict__ gw, const float* __restrict__ deg,
                              const float* __restrict__ bias, float* __restrict__ gsc,
                              int total) {
    int idx = blockIdx.x * blockDim.x + threadIdx.x;
    if (idx >= total) return;
    int n = idx >> 8, c = idx & 255;
    gsc[idx] = bias[c] + gw[idx] / deg[n];     // enorm(self) = 1/deg
}

// GCN edge scatter: one block per edge, thread per channel
__global__ __launch_bounds__(256) void gcn_edge(const int* __restrict__ src,
                                                const int* __restrict__ dst,
                                                const float* __restrict__ deg,
                                                const float* __restrict__ gw,
                                                float* __restrict__ gsc) {
    int e = blockIdx.x, c = threadIdx.x;
    int s = src[e], d = dst[e];
    float w = rsqrtf(deg[s]) * rsqrtf(deg[d]);
    atomicAdd(&gsc[(size_t)d * HID + c], gw[(size_t)s * HID + c] * w);
}

// BatchNorm (eval) + optional ReLU; emits f32 + bf16 (next layer operand)
__global__ void bn_act(const float* __restrict__ gsc,
                       const float* __restrict__ bg, const float* __restrict__ bb,
                       const float* __restrict__ bm, const float* __restrict__ bv,
                       float* __restrict__ gout, u16* __restrict__ gb,
                       int total, int relu) {
    int idx = blockIdx.x * blockDim.x + threadIdx.x;
    if (idx >= total) return;
    int c = idx & 255;
    float y = (gsc[idx] - bm[c]) * rsqrtf(bv[c] + EPS_BN) * bg[c] + bb[c];
    if (relu) y = fmaxf(y, 0.f);
    gout[idx] = y;
    gb[idx]   = f2bf(y);
}

// final: row L2 normalize -> g out, and pooled sum into xpool
__global__ __launch_bounds__(256) void final_norm_pool(const float* __restrict__ Gm,
                                                       const int* __restrict__ batch,
                                                       float* __restrict__ gout,
                                                       float* __restrict__ pool,
                                                       int nrows) {
    int row = blockIdx.x * 8 + (threadIdx.x >> 5);
    if (row >= nrows) return;
    int lane = threadIdx.x & 31;
    float y[8], ss = 0.f;
#pragma unroll
    for (int i = 0; i < 8; ++i) { y[i] = Gm[(size_t)row * HID + lane + 32 * i]; ss += y[i] * y[i]; }
    float nrm = fmaxf(sqrtf(wave_sum(ss)), 1e-12f);
    int grp = batch[row];
#pragma unroll
    for (int i = 0; i < 8; ++i) {
        float v = y[i] / nrm;
        gout[(size_t)row * HID + lane + 32 * i] = v;
        atomicAdd(&pool[(size_t)grp * HID + lane + 32 * i], v);
    }
}

// =====================================================================
extern "C" void kernel_launch(void* const* d_in, const int* in_sizes, int n_in,
                              void* d_out, int out_size, void* d_ws, size_t ws_size,
                              hipStream_t stream) {
    // ---- inputs (setup_inputs order) ----
    const float* x      = (const float*)d_in[0];
    const int*   eidx   = (const int*)d_in[1];
    const int*   batch  = (const int*)d_in[2];
    /* beta d_in[3] unused */
    const float* fc_W   = (const float*)d_in[4];
    const float* fc_b   = (const float*)d_in[5];
    const float* ln0_g  = (const float*)d_in[6];
    const float* ln0_b  = (const float*)d_in[7];
    const float* Wq     = (const float*)d_in[8];
    const float* bq     = (const float*)d_in[9];
    const float* Wk     = (const float*)d_in[10];
    const float* bk     = (const float*)d_in[11];
    const float* Wv     = (const float*)d_in[12];
    const float* bv     = (const float*)d_in[13];
    const float* ln1_g  = (const float*)d_in[14];
    const float* ln1_b  = (const float*)d_in[15];
    const float* gcn_W0 = (const float*)d_in[16];
    const float* gcn_b0 = (const float*)d_in[17];
    const float* gcn_W  = (const float*)d_in[18];
    const float* gcn_b  = (const float*)d_in[19];
    const float* bn_g   = (const float*)d_in[20];
    const float* bn_b   = (const float*)d_in[21];
    const float* bn_m   = (const float*)d_in[22];
    const float* bn_v   = (const float*)d_in[23];

    const int* e_src = eidx;
    const int* e_dst = eidx + E_EDGES;

    // ---- workspace layout ----
    char* W = (char*)d_ws;
    size_t off = 0;
    auto take = [&](size_t bytes) { size_t o = off; off += (bytes + 255) & ~(size_t)255; return o; };
    float* f_h    = (float*)(W + take((size_t)N_NODES * HID * 4));
    float* f_qkv  = (float*)(W + take((size_t)N_NODES * 768 * 4));
    float* f_t    = (float*)(W + take((size_t)N_NODES * HID * 4));
    u16*   b_xb   = (u16*)  (W + take((size_t)N_NODES * F_IN * 2));
    u16*   b_hb   = (u16*)  (W + take((size_t)N_NODES * HID * 2));
    u16*   b_qb   = (u16*)  (W + take((size_t)N_NODES * HID * 2));
    u16*   b_kT   = (u16*)  (W + take((size_t)HID * N_PAD * 2));
    u16*   b_vT   = (u16*)  (W + take((size_t)HID * N_PAD * 2));
    u16*   b_kvsT = (u16*)  (W + take((size_t)HID * HID * 2));
    u16*   b_fcWt = (u16*)  (W + take((size_t)HID * F_IN * 2));
    u16*   b_qkvWt= (u16*)  (W + take((size_t)768 * HID * 2));
    u16*   b_g0Wt = (u16*)  (W + take((size_t)HID * F_IN * 2));
    u16*   b_gWt  = (u16*)  (W + take((size_t)4 * HID * HID * 2));
    float* f_kvs  = (float*)(W + take((size_t)HID * HID * 4));
    float* f_small= (float*)(W + take(768 * 4));
    float* f_bqkv = (float*)(W + take(768 * 4));
    float* f_deg  = (float*)(W + take((size_t)N_NODES * 4));
    // aliases (lifetimes don't overlap)
    float* f_gw  = f_qkv;   // GCN GEMM output
    float* f_gsc = f_t;     // GCN scatter accumulator
    float* f_g   = f_h;     // GCN BN output (f32)
    u16*   b_gb  = b_hb;    // GCN next-layer bf16 operand

    float* out    = (float*)d_out;
    float* xpool  = out;                                   // [G,HID]
    float* g_out  = out + (size_t)G_GR * HID;              // [N,HID]
    float* xt_out = g_out + (size_t)N_NODES * HID;         // [N,HID]

    const int TPB = 256;
    auto blocks = [](long long n, int t) { return (unsigned)((n + t - 1) / t); };

    // ---- weight prep (transpose + bf16) ----
    transpose_bf16<<<blocks(F_IN * HID, TPB), TPB, 0, stream>>>(fc_W, b_fcWt, F_IN, HID);
    transpose_bf16<<<blocks(HID * HID, TPB), TPB, 0, stream>>>(Wq, b_qkvWt + 0 * HID * HID, HID, HID);
    transpose_bf16<<<blocks(HID * HID, TPB), TPB, 0, stream>>>(Wk, b_qkvWt + 1 * HID * HID, HID, HID);
    transpose_bf16<<<blocks(HID * HID, TPB), TPB, 0, stream>>>(Wv, b_qkvWt + 2 * HID * HID, HID, HID);
    transpose_bf16<<<blocks(F_IN * HID, TPB), TPB, 0, stream>>>(gcn_W0, b_g0Wt, F_IN, HID);
    for (int i = 0; i < 4; ++i)
        transpose_bf16<<<blocks(HID * HID, TPB), TPB, 0, stream>>>(
            gcn_W + (size_t)i * HID * HID, b_gWt + (size_t)i * HID * HID, HID, HID);
    pack_bias3<<<3, 256, 0, stream>>>(bq, bk, bv, f_bqkv);
    convert_bf16<<<blocks((long long)N_NODES * F_IN, TPB), TPB, 0, stream>>>(x, b_xb, N_NODES * F_IN);

    // ---- per-call accumulator init ----
    zero_f32<<<blocks(768, TPB), TPB, 0, stream>>>(f_small, 768);
    zero_f32<<<blocks(G_GR * HID, TPB), TPB, 0, stream>>>(xpool, G_GR * HID);
    init_deg<<<blocks(N_NODES, TPB), TPB, 0, stream>>>(f_deg, N_NODES);
    edge_deg<<<blocks(E_EDGES, TPB), TPB, 0, stream>>>(e_dst, f_deg, E_EDGES);

    dim3 gemm_grid_256(HID / 64, (N_NODES + 127) / 128);
    dim3 gemm_grid_768(768 / 64, (N_NODES + 127) / 128);

    // ---- TransConv branch ----
    wmma_gemm<<<gemm_grid_256, 256, 0, stream>>>(b_xb, b_fcWt, fc_b, f_t, N_NODES, F_IN, HID);
    ln_relu_rows<<<N_NODES / 8, 256, 0, stream>>>(f_t, ln0_g, ln0_b, f_h, b_hb, N_NODES);
    wmma_gemm<<<gemm_grid_768, 256, 0, stream>>>(b_hb, b_qkvWt, f_bqkv, f_qkv, N_NODES, HID, 768);
    qkv_post<<<N_PAD, 256, 0, stream>>>(f_qkv, b_qb, b_kT, b_vT, f_small, N_NODES, N_PAD);
    kvs_gemm<<<dim3(HID / 16, HID / 64), 256, 0, stream>>>(b_kT, b_vT, f_kvs, N_PAD);
    kvs_prep<<<HID, HID, 0, stream>>>(f_kvs, f_small, b_kvsT);
    wmma_gemm<<<gemm_grid_256, 256, 0, stream>>>(b_qb, b_kvsT, nullptr, f_t, N_NODES, HID, HID);
    attn_epilogue<<<N_NODES / 8, 256, 0, stream>>>(f_t, f_qkv, f_h, f_small, ln1_g, ln1_b, xt_out, N_NODES);

    // ---- GCN stack (f_qkv/f_t/f_h/b_hb reused from here on) ----
    const u16* Ain = b_xb; int Kin = F_IN;
    for (int i = 0; i < NUM_GC; ++i) {
        const u16* Wt = (i == 0) ? b_g0Wt : (b_gWt + (size_t)(i - 1) * HID * HID);
        const float* bias = (i == 0) ? gcn_b0 : (gcn_b + (size_t)(i - 1) * HID);
        wmma_gemm<<<gemm_grid_256, 256, 0, stream>>>(Ain, Wt, nullptr, f_gw, N_NODES, Kin, HID);
        gcn_self_init<<<blocks((long long)N_NODES * HID, TPB), TPB, 0, stream>>>(
            f_gw, f_deg, bias, f_gsc, N_NODES * HID);
        gcn_edge<<<E_EDGES, 256, 0, stream>>>(e_src, e_dst, f_deg, f_gw, f_gsc);
        bn_act<<<blocks((long long)N_NODES * HID, TPB), TPB, 0, stream>>>(
            f_gsc, bn_g + (size_t)i * HID, bn_b + (size_t)i * HID,
            bn_m + (size_t)i * HID, bn_v + (size_t)i * HID,
            f_g, b_gb, N_NODES * HID, (i != NUM_GC - 1) ? 1 : 0);
        Ain = b_gb; Kin = HID;
    }
    final_norm_pool<<<N_NODES / 8, 256, 0, stream>>>(f_g, batch, g_out, xpool, N_NODES);
    (void)in_sizes; (void)n_in; (void)out_size; (void)ws_size;
}
